// HGT_40132174414160
// MI455X (gfx1250) — matrix operand order repaired and checked by reading, not compiled, not run on hardware
//
#include <hip/hip_runtime.h>

#define N_NODES 50000
#define N_EDGES 800000
#define N_PAIRS 100000
#define HID 128
#define NHEAD 8
#define DH 16
#define NLAYER 3

typedef __attribute__((ext_vector_type(16))) __bf16 v16bf;
typedef __attribute__((ext_vector_type(8)))  float    v8f;
typedef __attribute__((ext_vector_type(4)))  unsigned v4u;

union ABfrag { v16bf v; unsigned u[8]; v4u q[2]; };

__device__ __forceinline__ unsigned short f2bf(float f) {
  unsigned u = __float_as_uint(f);
  u += 0x7fffu + ((u >> 16) & 1u);          // round-to-nearest-even
  return (unsigned short)(u >> 16);
}
__device__ __forceinline__ unsigned fenc(float f) {   // order-preserving float->uint
  unsigned u = __float_as_uint(f);
  return (u & 0x80000000u) ? ~u : (u | 0x80000000u);
}
__device__ __forceinline__ float fdec(unsigned u) {
  return (u & 0x80000000u) ? __uint_as_float(u ^ 0x80000000u) : __uint_as_float(~u);
}
__device__ __forceinline__ float lrelu(float x) { return x > 0.f ? x : 0.2f * x; }

// ---- fold per-head [16x16] transform into the 128x128 weight ------------
// Weff[i, h*16+e] = sum_d Wsrc[i, h*16+d] * T[h, d, e]
__global__ __launch_bounds__(256)
void make_eff(const float* __restrict__ Wsrc, const float* __restrict__ T,
              float* __restrict__ Weff) {
  int idx = blockIdx.x * 256 + threadIdx.x;           // 16384
  int i = idx >> 7, c = idx & 127;
  int hh = c >> 4, e = c & 15;
  float acc = 0.f;
#pragma unroll
  for (int d = 0; d < 16; ++d)
    acc += Wsrc[i * HID + hh * 16 + d] * T[hh * 256 + d * 16 + e];
  Weff[idx] = acc;
}

// ---- pre-swizzle f32 [K=128, N=128] weight into WMMA B-fragment layout ---
// out[((ct*4+kt)*32 + lane)*16 + e] = bf16(W[K,n]); K = kt*32 + 16*(lane>>4) + e
__global__ __launch_bounds__(256)
void pack_b(const float* __restrict__ W, unsigned short* __restrict__ out) {
  int idx = blockIdx.x * 256 + threadIdx.x;           // 16384
  int e    = idx & 15;
  int lane = (idx >> 4) & 31;
  int kt   = (idx >> 9) & 3;
  int ct   = idx >> 11;
  int K = kt * 32 + ((lane >> 4) << 4) + e;
  int n = ct * 16 + (lane & 15);
  out[idx] = f2bf(W[K * HID + n]);
}

// ---- main GEMM: Y = epilogue( X[rows,128] @ W[128,128] ) -----------------
// block = 256 thr (8 waves); block covers 16 rows; wave w -> 16-col tile w.
// A staged fp32->bf16 in LDS (stride 130 -> conflict-free), B preswizzled.
__global__ __launch_bounds__(256)
void gemm128_wmma(const float* __restrict__ X,
                  const unsigned short* __restrict__ Wp,
                  const float* __restrict__ bias,
                  const float* __restrict__ skip_old,
                  const float* __restrict__ gatep,
                  int relu, float* __restrict__ Y) {
  __shared__ __align__(16) unsigned short As[16 * 130];
  int rowBase = blockIdx.x * 16;
  {   // cooperative fill: 2048 f32 -> bf16, 8 per thread
    int t = threadIdx.x;
    int r = (t * 8) >> 7;
    int c = (t * 8) & 127;
    const float4* src = (const float4*)(X + (size_t)(rowBase + r) * HID + c);
    float4 f0 = src[0], f1 = src[1];
    unsigned* dst = (unsigned*)&As[r * 130 + c];
    dst[0] = (unsigned)f2bf(f0.x) | ((unsigned)f2bf(f0.y) << 16);
    dst[1] = (unsigned)f2bf(f0.z) | ((unsigned)f2bf(f0.w) << 16);
    dst[2] = (unsigned)f2bf(f1.x) | ((unsigned)f2bf(f1.y) << 16);
    dst[3] = (unsigned)f2bf(f1.z) | ((unsigned)f2bf(f1.w) << 16);
  }
  __syncthreads();
  int lane = threadIdx.x & 31;
  int w    = threadIdx.x >> 5;
  int hi   = lane >> 4;
  int m    = lane & 15;
  v8f acc = {};
#pragma unroll
  for (int kt = 0; kt < 4; ++kt) {
    ABfrag a, b;
#pragma unroll
    for (int j = 0; j < 8; ++j) {
      // A layout (matches packed B): VGPR j pair K0 = 16*hi + 2*j
      int K0 = kt * 32 + (hi << 4) + (j << 1);
      a.u[j] = *(const unsigned*)&As[m * 130 + K0];
    }
    const v4u* bp = (const v4u*)(Wp + (((w * 4 + kt) * 32 + lane) << 4));
    b.q[0] = bp[0]; b.q[1] = bp[1];
    acc = __builtin_amdgcn_wmma_f32_16x16x32_bf16(
        false, a.v, false, b.v, (short)0, acc, false, false);
  }
  int n   = lane & 15;
  int col = (w << 4) + n;
  float bval = bias ? bias[col] : 0.f;
  float ga = 1.f, gb = 0.f;
  if (gatep) { float s = 1.f / (1.f + __expf(-gatep[0])); ga = s; gb = 1.f - s; }
#pragma unroll
  for (int r = 0; r < 8; ++r) {                  // C layout: M = 8*hi + r
    int row = rowBase + (hi << 3) + r;
    size_t off = (size_t)row * HID + col;
    float v = acc[r] + bval;
    if (skip_old) v = ga * v + gb * skip_old[off];
    if (relu)     v = fmaxf(v, 0.f);
    Y[off] = v;
  }
}

// ---- per-layer scratch init ---------------------------------------------
__global__ __launch_bounds__(256)
void zero_layer(float* agg, float* denom, unsigned* amax) {
  int i = blockIdx.x * 256 + threadIdx.x;        // N*128
  agg[i] = 0.f;
  if (i < N_NODES * NHEAD) { denom[i] = 0.f; amax[i] = 0u; }
}

// ---- edge pass A: logits + segment max ----------------------------------
__global__ __launch_bounds__(256)
void edge_logits(const int* __restrict__ esrc, const int* __restrict__ edst,
                 const float* __restrict__ kw, const float* __restrict__ q,
                 const float* __restrict__ pri_l,
                 float* __restrict__ logits, unsigned* __restrict__ amax) {
  int e = blockIdx.x * 256 + threadIdx.x;
  if (e >= N_EDGES) return;
  int s = esrc[e], dd = edst[e];
  const float4* kp = (const float4*)(kw + (size_t)s * HID);
  const float4* qp = (const float4*)(q + (size_t)dd * HID);
#pragma unroll
  for (int h = 0; h < NHEAD; ++h) {
    float acc = 0.f;
#pragma unroll
    for (int t = 0; t < 4; ++t) {
      float4 a4 = kp[h * 4 + t], b4 = qp[h * 4 + t];
      acc += a4.x * b4.x + a4.y * b4.y + a4.z * b4.z + a4.w * b4.w;
    }
    float lg = acc * pri_l[h] * 0.25f;           // pri/sqrt(16)
    logits[(size_t)e * NHEAD + h] = lg;
    atomicMax(amax + dd * NHEAD + h, fenc(lg));
  }
}

// ---- edge pass B: exp + segment sum -------------------------------------
__global__ __launch_bounds__(256)
void edge_softmax(const int* __restrict__ edst, float* __restrict__ logits,
                  const unsigned* __restrict__ amax, float* __restrict__ denom) {
  int i = blockIdx.x * 256 + threadIdx.x;        // E*8
  int e = i >> 3, h = i & 7;
  int dd = edst[e];
  float ex = __expf(logits[i] - fdec(amax[dd * NHEAD + h]));
  logits[i] = ex;
  unsafeAtomicAdd(denom + dd * NHEAD + h, ex);
}

// ---- edge pass C: normalize + weighted scatter-sum ----------------------
__global__ __launch_bounds__(256)
void edge_scatter(const int* __restrict__ esrc, const int* __restrict__ edst,
                  const float* __restrict__ ew, const float* __restrict__ logits,
                  const float* __restrict__ denom, const float* __restrict__ mv,
                  float* __restrict__ agg) {
  int e = blockIdx.x * 256 + threadIdx.x;
  if (e >= N_EDGES) return;
  int s = esrc[e], dd = edst[e];
  float c0 = ew[e];
  const float4* mp = (const float4*)(mv + (size_t)s * HID);
#pragma unroll
  for (int h = 0; h < NHEAD; ++h) {
    float attn = logits[(size_t)e * NHEAD + h] / denom[dd * NHEAD + h];
    float c = c0 * attn;
    float* ag = agg + (size_t)dd * HID + h * 16;
#pragma unroll
    for (int t = 0; t < 4; ++t) {
      float4 m4 = mp[h * 4 + t];
      unsafeAtomicAdd(ag + t * 4 + 0, m4.x * c);
      unsafeAtomicAdd(ag + t * 4 + 1, m4.y * c);
      unsafeAtomicAdd(ag + t * 4 + 2, m4.z * c);
      unsafeAtomicAdd(ag + t * 4 + 3, m4.w * c);
    }
  }
}

// ---- link predictor MLP: one 64-thread block per pair -------------------
__global__ __launch_bounds__(64)
void predictor_kernel(const float* __restrict__ h,
                      const int* __restrict__ pos_src, const int* __restrict__ pos_dst,
                      const int* __restrict__ neg_src, const int* __restrict__ neg_dst,
                      const float* __restrict__ W1, const float* __restrict__ b1,
                      const float* __restrict__ W2, const float* __restrict__ b2,
                      const float* __restrict__ W3, const float* __restrict__ b3,
                      float* __restrict__ out) {
  __shared__ float z[HID];
  __shared__ float z1[64];
  __shared__ float z2[32];
  int row = blockIdx.x;
  const int *srcs, *dsts; float* o;
  if (row < N_PAIRS) { srcs = pos_src; dsts = pos_dst; o = out + row; }
  else { row -= N_PAIRS; srcs = neg_src; dsts = neg_dst; o = out + N_PAIRS + row; }
  int s = srcs[row], d = dsts[row];
  int t = threadIdx.x;
  z[t]      = h[(size_t)s * HID + t]      * h[(size_t)d * HID + t];
  z[t + 64] = h[(size_t)s * HID + t + 64] * h[(size_t)d * HID + t + 64];
  __syncthreads();
  float acc = b1[t];
#pragma unroll 8
  for (int k = 0; k < HID; ++k) acc += z[k] * W1[k * 64 + t];
  z1[t] = lrelu(acc);
  __syncthreads();
  if (t < 32) {
    float a2 = b2[t];
#pragma unroll 8
    for (int k = 0; k < 64; ++k) a2 += z1[k] * W2[k * 32 + t];
    z2[t] = lrelu(a2);
  }
  __syncthreads();
  if (t == 0) {
    float a3 = b3[0];
#pragma unroll
    for (int k = 0; k < 32; ++k) a3 += z2[k] * W3[k];
    *o = a3;
  }
}

__global__ __launch_bounds__(256)
void copy_f(const float* __restrict__ src, float* __restrict__ dst) {
  int i = blockIdx.x * 256 + threadIdx.x;
  dst[i] = src[i];
}

extern "C" void kernel_launch(void* const* d_in, const int* in_sizes, int n_in,
                              void* d_out, int out_size, void* d_ws, size_t ws_size,
                              hipStream_t stream) {
  const float* x    = (const float*)d_in[0];
  const int*   esrc = (const int*)d_in[1];
  const int*   edst = (const int*)d_in[2];
  const float* ew   = (const float*)d_in[3];
  const int*   psrc = (const int*)d_in[4];
  const int*   pdst = (const int*)d_in[5];
  const int*   nsrc = (const int*)d_in[6];
  const int*   ndst = (const int*)d_in[7];
  const float* W_in = (const float*)d_in[8];
  const float* b_in = (const float*)d_in[9];
  const float* Wk   = (const float*)d_in[10];
  const float* Wq   = (const float*)d_in[11];
  const float* Wv   = (const float*)d_in[12];
  const float* attw = (const float*)d_in[13];
  const float* msgw = (const float*)d_in[14];
  const float* pri  = (const float*)d_in[15];
  const float* Wa   = (const float*)d_in[16];
  const float* skip = (const float*)d_in[17];
  const float* W1   = (const float*)d_in[18];
  const float* b1   = (const float*)d_in[19];
  const float* W2   = (const float*)d_in[20];
  const float* b2   = (const float*)d_in[21];
  const float* W3   = (const float*)d_in[22];
  const float* b3   = (const float*)d_in[23];
  float* out = (float*)d_out;

  char* wsp = (char*)d_ws;
  auto carve = [&](size_t bytes) {
    char* p = wsp; wsp += (bytes + 255) & ~(size_t)255; return p;
  };
  float*    h      = (float*)carve((size_t)N_NODES * HID * 4);
  float*    kw     = (float*)carve((size_t)N_NODES * HID * 4);
  float*    qb     = (float*)carve((size_t)N_NODES * HID * 4);
  float*    mv     = (float*)carve((size_t)N_NODES * HID * 4);
  float*    agg    = (float*)carve((size_t)N_NODES * HID * 4);
  float*    logits = (float*)carve((size_t)N_EDGES * NHEAD * 4);
  float*    denom  = (float*)carve((size_t)N_NODES * NHEAD * 4);
  unsigned* amax   = (unsigned*)carve((size_t)N_NODES * NHEAD * 4);
  float*    wtmp   = (float*)carve((size_t)HID * HID * 4);
  unsigned short* packs = (unsigned short*)carve((size_t)13 * 16384 * 2);
  auto pW = [&](int i) { return packs + (size_t)i * 16384; };

  dim3 b256(256), b64(64);
  // ---- weight prep: fold per-head transforms, pre-swizzle to WMMA-B ----
  pack_b<<<64, b256, 0, stream>>>(W_in, pW(0));
  for (int l = 0; l < NLAYER; ++l) {
    pack_b<<<64, b256, 0, stream>>>(Wq + (size_t)l * HID * HID, pW(1 + 4 * l + 0));
    make_eff<<<64, b256, 0, stream>>>(Wk + (size_t)l * HID * HID,
                                      attw + (size_t)l * NHEAD * DH * DH, wtmp);
    pack_b<<<64, b256, 0, stream>>>(wtmp, pW(1 + 4 * l + 1));
    make_eff<<<64, b256, 0, stream>>>(Wv + (size_t)l * HID * HID,
                                      msgw + (size_t)l * NHEAD * DH * DH, wtmp);
    pack_b<<<64, b256, 0, stream>>>(wtmp, pW(1 + 4 * l + 2));
    pack_b<<<64, b256, 0, stream>>>(Wa + (size_t)l * HID * HID, pW(1 + 4 * l + 3));
  }

  const int gRows  = N_NODES / 16;            // 3125
  const int gElems = N_NODES * HID / 256;     // 25000
  const int gEdges = (N_EDGES + 255) / 256;   // 3125
  const int gEH    = N_EDGES * NHEAD / 256;   // 25000

  // input projection: h = relu(x @ W_in + b_in)
  gemm128_wmma<<<gRows, b256, 0, stream>>>(x, pW(0), b_in, nullptr, nullptr, 1, h);

  for (int l = 0; l < NLAYER; ++l) {
    zero_layer<<<gElems, b256, 0, stream>>>(agg, denom, amax);
    gemm128_wmma<<<gRows, b256, 0, stream>>>(h, pW(1 + 4 * l + 1), nullptr, nullptr, nullptr, 0, kw);
    gemm128_wmma<<<gRows, b256, 0, stream>>>(h, pW(1 + 4 * l + 0), nullptr, nullptr, nullptr, 0, qb);
    gemm128_wmma<<<gRows, b256, 0, stream>>>(h, pW(1 + 4 * l + 2), nullptr, nullptr, nullptr, 0, mv);
    edge_logits<<<gEdges, b256, 0, stream>>>(esrc, edst, kw, qb, pri + l * NHEAD, logits, amax);
    edge_softmax<<<gEH, b256, 0, stream>>>(edst, logits, amax, denom);
    edge_scatter<<<gEdges, b256, 0, stream>>>(esrc, edst, ew, logits, denom, mv, agg);
    // h = gate * (agg @ Wa) + (1-gate) * h ; relu except last layer
    gemm128_wmma<<<gRows, b256, 0, stream>>>(agg, pW(1 + 4 * l + 3), nullptr, h,
                                             skip + l, (l < NLAYER - 1) ? 1 : 0, h);
  }

  copy_f<<<gElems, b256, 0, stream>>>(h, out + 2 * (size_t)N_PAIRS);
  predictor_kernel<<<2 * N_PAIRS, b64, 0, stream>>>(h, psrc, pdst, nsrc, ndst,
                                                    W1, b1, W2, b2, W3, b3, out);
}
